// ReadUnit_34102040330305
// MI455X (gfx1250) — matrix operand order repaired and checked by reading, not compiled, not run on hardware
//
#include <hip/hip_runtime.h>
#include <hip/hip_bf16.h>

// ---------------------------------------------------------------------------
// ReadUnit fused op for MI455X (gfx1250, wave32, WMMA).
// B=64, H=W=14, D=1024.  All GEMMs in bf16 WMMA (f32 accumulate).
// Round 3: 32-row M-blocks (each B fragment feeds 2 WMMAs -> halves L2
// weight traffic), async-to-LDS k staging, packed bf16 conversion epilogue.
// ---------------------------------------------------------------------------

typedef __attribute__((ext_vector_type(16))) __bf16 v16bf;
typedef __attribute__((ext_vector_type(8)))  float  v8f;

#define D_DIM   1024
#define HW      196
#define BATCH   64
#define ROWS    (BATCH * HW)      /* 12544 */
#define MROWS   32                /* rows per workgroup */
#define MBLKS   (ROWS / MROWS)    /* 392   */

// ---- bf16 helpers ----------------------------------------------------------
__device__ __forceinline__ unsigned short f2b(float x) {
    union { float f; unsigned int u; } c; c.f = x;
    unsigned int u = c.u;
    u += 0x7FFFu + ((u >> 16) & 1u);          // round-to-nearest-even
    return (unsigned short)(u >> 16);
}
__device__ __forceinline__ float b2f(unsigned short h) {
    union { unsigned int u; float f; } c; c.u = ((unsigned int)h) << 16;
    return c.f;
}

// Convert two f32 -> two bf16 (hardware packed cvt when available) and store
// to two (non-adjacent) LDS locations.
__device__ __forceinline__ void cvt2_store(unsigned short* p0, unsigned short* p1,
                                           float x0, float x1) {
#if __has_builtin(__builtin_amdgcn_cvt_pk_bf16_f32)
    typedef __attribute__((ext_vector_type(2))) __bf16 v2bf;
    union { v2bf v; unsigned short s[2]; } u;
    u.v = __builtin_amdgcn_cvt_pk_bf16_f32(x0, x1);
    *p0 = u.s[0];
    *p1 = u.s[1];
#else
    *p0 = f2b(x0);
    *p1 = f2b(x1);
#endif
}

// ---- WMMA fragment helpers -------------------------------------------------
union FragU { uint4 q[2]; v16bf v; };

// A fragment (16x32 bf16): lane = 16*khalf + m.
__device__ __forceinline__ v16bf load_a(const unsigned short* base /* row m start */,
                                        int kb, int khalf) {
    FragU f;
    f.q[0] = *(const uint4*)(base + kb + khalf * 16);
    f.q[1] = *(const uint4*)(base + kb + khalf * 16 + 8);
    return f.v;
}

// B fragment (32x16 bf16): lane = 16*khalf + n; 32 contiguous bytes of row n.
__device__ __forceinline__ v16bf load_b(const unsigned short* wrow /* row n start */,
                                        int kb, int khalf) {
    const uint4* p = (const uint4*)(wrow + kb + khalf * 16);
    FragU f;
    f.q[0] = p[0];
    f.q[1] = p[1];
    return f.v;
}

__device__ __forceinline__ v8f wmma_bf16(v16bf a, v16bf b, v8f c) {
    return __builtin_amdgcn_wmma_f32_16x16x32_bf16(
        false, a, false, b, (short)0, c, false, false);
}

#define V8F_ZERO (v8f){0.f, 0.f, 0.f, 0.f, 0.f, 0.f, 0.f, 0.f}

// ---------------------------------------------------------------------------
// f32 -> bf16 conversion kernel
// ---------------------------------------------------------------------------
__global__ void cvt_bf16_kernel(const float* __restrict__ src,
                                unsigned short* __restrict__ dst, int n) {
    int i = blockIdx.x * blockDim.x + threadIdx.x;
    if (i < n) dst[i] = f2b(src[i]);
}

// ---------------------------------------------------------------------------
// Wm = m_i_1 @ W_ddm^T + b_dm   -> f32 [64,1024].  1 wave per 16x16 tile.
// ---------------------------------------------------------------------------
__global__ __launch_bounds__(32) void wm_kernel(
    const unsigned short* __restrict__ m_bf,
    const unsigned short* __restrict__ wddm_bf,
    const float* __restrict__ b_dm,
    float* __restrict__ Wm)
{
    int mt = blockIdx.x & 3;
    int nt = blockIdx.x >> 2;
    int lane  = threadIdx.x;
    int lo    = lane & 15;
    int khalf = lane >> 4;

    const unsigned short* arow = m_bf + (size_t)(mt * 16 + lo) * D_DIM;
    const unsigned short* brow = wddm_bf + (size_t)(nt * 16 + lo) * D_DIM;

    v8f acc = V8F_ZERO;
    for (int kb = 0; kb < D_DIM; kb += 32)
        acc = wmma_bf16(load_a(arow, kb, khalf), load_b(brow, kb, khalf), acc);

    int n = nt * 16 + lo;
    float bias = b_dm[n];
#pragma unroll
    for (int v = 0; v < 8; ++v) {
        int row = mt * 16 + v + 8 * khalf;
        Wm[(size_t)row * D_DIM + n] = acc[v] + bias;
    }
}

// ---------------------------------------------------------------------------
// Fused main kernel: one workgroup per 32-row M-block, 8 waves (wave32).
// Each wave: 8 N-tiles x 2 M-tiles = 16 accumulators; every B fragment is
// reused by 2 WMMAs; A fragments loaded once per K-step from LDS.
// ---------------------------------------------------------------------------
__global__ __launch_bounds__(256, 1) void fused_kernel(
    const float* __restrict__ c_i,
    const float* __restrict__ b_dk,
    const float* __restrict__ b_d1,
    const float* __restrict__ b_d2,
    const unsigned short* __restrict__ wddk_bf,
    const unsigned short* __restrict__ wd2d_bf,
    const unsigned short* __restrict__ wdd_bf,
    const unsigned short* __restrict__ k_bf,
    const float* __restrict__ Wm,
    float* __restrict__ out)
{
    __shared__ unsigned short kt[MROWS * D_DIM];   // 64 KB  k-tile (bf16)
    __shared__ unsigned short i1[MROWS * D_DIM];   // 64 KB  intermediate / stage3 out
    __shared__ unsigned short i2[MROWS * D_DIM];   // 64 KB  intermediate

    const int row0 = blockIdx.x * MROWS;
    const int tid  = threadIdx.x;
    const int wave = tid >> 5;
    const int lane = tid & 31;
    const int lo    = lane & 15;
    const int khalf = lane >> 4;
    const int nbase = wave * 8;

    // ---- stage k-tile into LDS: async DMA (ASYNCcnt path), 16B per lane ----
    {
        const unsigned long long gbase =
            (unsigned long long)(size_t)(k_bf + (size_t)row0 * D_DIM);
        const unsigned ldsbase = (unsigned)(size_t)&kt[0];
#pragma unroll 1
        for (int i = tid; i < MROWS * D_DIM / 8; i += 256) {
            unsigned long long ga = gbase + (unsigned long long)i * 16ull;
            unsigned           la = ldsbase + (unsigned)i * 16u;
            asm volatile("global_load_async_to_lds_b128 %0, %1, off"
                         :: "v"(la), "v"(ga) : "memory");
        }
        asm volatile("s_wait_asynccnt 0x0" ::: "memory");
    }
    __syncthreads();

    const unsigned short* a_kt0 = kt + (size_t)lo * D_DIM;
    const unsigned short* a_kt1 = kt + (size_t)(16 + lo) * D_DIM;
    const unsigned short* a_i10 = i1 + (size_t)lo * D_DIM;
    const unsigned short* a_i11 = i1 + (size_t)(16 + lo) * D_DIM;
    const unsigned short* a_i20 = i2 + (size_t)lo * D_DIM;
    const unsigned short* a_i21 = i2 + (size_t)(16 + lo) * D_DIM;

    // =================== stage 1: i1 = bf16((kt@Wddk^T + b_dk)*Wm) ==========
    {
        const unsigned short* brow[8];
        v8f acc0[8], acc1[8];
#pragma unroll
        for (int t = 0; t < 8; ++t) {
            acc0[t] = V8F_ZERO;
            acc1[t] = V8F_ZERO;
            brow[t] = wddk_bf + (size_t)((nbase + t) * 16 + lo) * D_DIM;
            __builtin_prefetch(brow[t], 0, 3);
        }
#pragma unroll 1
        for (int kb = 0; kb < D_DIM; kb += 32) {
            v16bf a0 = load_a(a_kt0, kb, khalf);
            v16bf a1 = load_a(a_kt1, kb, khalf);
#pragma unroll
            for (int t = 0; t < 8; ++t) {
                v16bf b = load_b(brow[t], kb, khalf);
                acc0[t] = wmma_bf16(a0, b, acc0[t]);
                acc1[t] = wmma_bf16(a1, b, acc1[t]);
            }
        }
#pragma unroll
        for (int t = 0; t < 8; ++t) {
            const int n = (nbase + t) * 16 + lo;
            const float bias = b_dk[n];
#pragma unroll
            for (int v = 0; v < 8; ++v) {
                int mm0 = v + 8 * khalf, mm1 = mm0 + 16;
                int b0 = (row0 + mm0) / HW, b1 = (row0 + mm1) / HW;
                cvt2_store(&i1[mm0 * D_DIM + n], &i1[mm1 * D_DIM + n],
                           (acc0[t][v] + bias) * Wm[(size_t)b0 * D_DIM + n],
                           (acc1[t][v] + bias) * Wm[(size_t)b1 * D_DIM + n]);
            }
        }
    }
    __syncthreads();

    // =================== stage 2: i2 = bf16((i1@W1^T + kt@W2^T + b_d1)*c) ===
    {
        const unsigned short* brow1[8];
        const unsigned short* brow2[8];
        v8f acc0[8], acc1[8];
#pragma unroll
        for (int t = 0; t < 8; ++t) {
            acc0[t]  = V8F_ZERO;
            acc1[t]  = V8F_ZERO;
            brow1[t] = wd2d_bf + (size_t)((nbase + t) * 16 + lo) * (2 * D_DIM);
            brow2[t] = brow1[t] + D_DIM;
            __builtin_prefetch(brow1[t], 0, 3);
        }
#pragma unroll 1
        for (int kb = 0; kb < D_DIM; kb += 32) {
            v16bf a0 = load_a(a_i10, kb, khalf);
            v16bf a1 = load_a(a_i11, kb, khalf);
#pragma unroll
            for (int t = 0; t < 8; ++t) {
                v16bf b = load_b(brow1[t], kb, khalf);
                acc0[t] = wmma_bf16(a0, b, acc0[t]);
                acc1[t] = wmma_bf16(a1, b, acc1[t]);
            }
        }
#pragma unroll 1
        for (int kb = 0; kb < D_DIM; kb += 32) {
            v16bf a0 = load_a(a_kt0, kb, khalf);
            v16bf a1 = load_a(a_kt1, kb, khalf);
#pragma unroll
            for (int t = 0; t < 8; ++t) {
                v16bf b = load_b(brow2[t], kb, khalf);
                acc0[t] = wmma_bf16(a0, b, acc0[t]);
                acc1[t] = wmma_bf16(a1, b, acc1[t]);
            }
        }
#pragma unroll
        for (int t = 0; t < 8; ++t) {
            const int n = (nbase + t) * 16 + lo;
            const float bias = b_d1[n];
#pragma unroll
            for (int v = 0; v < 8; ++v) {
                int mm0 = v + 8 * khalf, mm1 = mm0 + 16;
                int b0 = (row0 + mm0) / HW, b1 = (row0 + mm1) / HW;
                cvt2_store(&i2[mm0 * D_DIM + n], &i2[mm1 * D_DIM + n],
                           (acc0[t][v] + bias) * c_i[(size_t)b0 * D_DIM + n],
                           (acc1[t][v] + bias) * c_i[(size_t)b1 * D_DIM + n]);
            }
        }
    }
    __syncthreads();

    // =================== stage 3: i1 = bf16(i2@Wdd^T + b_d2) ================
    {
        const unsigned short* brow[8];
        v8f acc0[8], acc1[8];
#pragma unroll
        for (int t = 0; t < 8; ++t) {
            acc0[t] = V8F_ZERO;
            acc1[t] = V8F_ZERO;
            brow[t] = wdd_bf + (size_t)((nbase + t) * 16 + lo) * D_DIM;
            __builtin_prefetch(brow[t], 0, 3);
        }
#pragma unroll 1
        for (int kb = 0; kb < D_DIM; kb += 32) {
            v16bf a0 = load_a(a_i20, kb, khalf);
            v16bf a1 = load_a(a_i21, kb, khalf);
#pragma unroll
            for (int t = 0; t < 8; ++t) {
                v16bf b = load_b(brow[t], kb, khalf);
                acc0[t] = wmma_bf16(a0, b, acc0[t]);
                acc1[t] = wmma_bf16(a1, b, acc1[t]);
            }
        }
#pragma unroll
        for (int t = 0; t < 8; ++t) {
            const int n = (nbase + t) * 16 + lo;
            const float bias = b_d2[n];
#pragma unroll
            for (int v = 0; v < 8; ++v) {
                int mm0 = v + 8 * khalf, mm1 = mm0 + 16;
                cvt2_store(&i1[mm0 * D_DIM + n], &i1[mm1 * D_DIM + n],
                           acc0[t][v] + bias, acc1[t][v] + bias);
            }
        }
    }
    __syncthreads();

    // =================== softmax + weighted spatial sum ======================
    // 8 waves x 4 rows each; wave32 shuffle reductions.
#pragma unroll 1
    for (int r = wave * 4; r < wave * 4 + 4; ++r) {
        const unsigned short* srow = i1 + r * D_DIM;
        const unsigned short* krow = kt + r * D_DIM;
        const int bat = (row0 + r) / HW;

        float mx = -3.0e38f;
        for (int c = lane; c < D_DIM; c += 32) mx = fmaxf(mx, b2f(srow[c]));
        for (int off = 16; off > 0; off >>= 1) mx = fmaxf(mx, __shfl_xor(mx, off, 32));

        float sum = 0.f;
        for (int c = lane; c < D_DIM; c += 32) sum += __expf(b2f(srow[c]) - mx);
        for (int off = 16; off > 0; off >>= 1) sum += __shfl_xor(sum, off, 32);
        const float inv = 1.0f / sum;

        for (int c = lane; c < D_DIM; c += 32) {
            float rv = __expf(b2f(srow[c]) - mx) * inv;
            float kv = b2f(krow[c]);
            atomicAdd(&out[(size_t)bat * D_DIM + c], rv * kv);
        }
    }
}

// ---------------------------------------------------------------------------
// Host-side launcher
// ---------------------------------------------------------------------------
extern "C" void kernel_launch(void* const* d_in, const int* in_sizes, int n_in,
                              void* d_out, int out_size, void* d_ws, size_t ws_size,
                              hipStream_t stream) {
    (void)in_sizes; (void)n_in; (void)out_size; (void)ws_size;

    const float* c_i   = (const float*)d_in[0];
    const float* m_i_1 = (const float*)d_in[1];
    const float* k_hw  = (const float*)d_in[2];
    const float* W_ddm = (const float*)d_in[3];
    const float* b_dm  = (const float*)d_in[4];
    const float* W_ddk = (const float*)d_in[5];
    const float* b_dk  = (const float*)d_in[6];
    const float* W_d2d = (const float*)d_in[7];
    const float* b_d1  = (const float*)d_in[8];
    const float* W_dd  = (const float*)d_in[9];
    const float* b_d2  = (const float*)d_in[10];
    float* out = (float*)d_out;

    // ---- workspace layout (bf16 mirrors + Wm f32) ----
    char* w = (char*)d_ws;
    size_t off = 0;
    unsigned short* m_bf    = (unsigned short*)(w + off); off += (size_t)BATCH * D_DIM * 2;
    unsigned short* wddm_bf = (unsigned short*)(w + off); off += (size_t)D_DIM * D_DIM * 2;
    unsigned short* wddk_bf = (unsigned short*)(w + off); off += (size_t)D_DIM * D_DIM * 2;
    unsigned short* wd2d_bf = (unsigned short*)(w + off); off += (size_t)D_DIM * 2 * D_DIM * 2;
    unsigned short* wdd_bf  = (unsigned short*)(w + off); off += (size_t)D_DIM * D_DIM * 2;
    unsigned short* k_bf    = (unsigned short*)(w + off); off += (size_t)ROWS * D_DIM * 2;
    float*          Wm      = (float*)(w + off);          off += (size_t)BATCH * D_DIM * 4;

    hipMemsetAsync(out, 0, (size_t)BATCH * D_DIM * sizeof(float), stream);

    auto cvt = [&](const float* s, unsigned short* d, int n) {
        cvt_bf16_kernel<<<(n + 255) / 256, 256, 0, stream>>>(s, d, n);
    };
    cvt(m_i_1, m_bf,    BATCH * D_DIM);
    cvt(W_ddm, wddm_bf, D_DIM * D_DIM);
    cvt(W_ddk, wddk_bf, D_DIM * D_DIM);
    cvt(W_d2d, wd2d_bf, D_DIM * 2 * D_DIM);
    cvt(W_dd,  wdd_bf,  D_DIM * D_DIM);
    cvt(k_hw,  k_bf,    ROWS * D_DIM);

    wm_kernel<<<4 * 64, 32, 0, stream>>>(m_bf, wddm_bf, b_dm, Wm);

    fused_kernel<<<MBLKS, 256, 0, stream>>>(
        c_i, b_dk, b_d1, b_d2, wddk_bf, wd2d_bf, wdd_bf, k_bf, Wm, out);
}
